// ConvolutionalDRAW_24824910971114
// MI455X (gfx1250) — matrix operand order repaired and checked by reading, not compiled, not run on hardware
//
#include <hip/hip_runtime.h>
#include <hip/hip_bf16.h>
#include <stdint.h>

// ---------------------------------------------------------------------------
// ConvDRAW for MI455X (gfx1250). All convolutions are implicit GEMMs executed
// with v_wmma_f32_16x16x32_f16 (fp16 A/B, fp32 accum). Compute-bound:
// ~3.3 TFLOP total vs ~hundreds of MB of traffic at 23.3 TB/s HBM.
// Conv kernel: 128x64 block tile, 8 wave32 (one 16-row strip each, 4 N-sub-
// tiles = 4 back-to-back WMMAs/K-step), double-buffered LDS with register
// staging so next tile's HBM loads + im2col gather overlap current WMMAs.
// ---------------------------------------------------------------------------

typedef _Float16 half_t;
typedef __attribute__((ext_vector_type(16))) _Float16 v16h;
typedef __attribute__((ext_vector_type(8)))  float    v8f;

#define NB      64          // batch
#define SP      256         // 16x16 spatial
#define NTOT    (NB * SP)   // 16384 GEMM columns
#define BM      128
#define BN      64
#define BK      32
#define APITCH  40          // halfs per LDS A row (80B, 16B aligned)
#define BPITCH  40          // halfs per LDS B row (transposed [n][k])

__device__ __forceinline__ float sigm(float x) { return 1.0f / (1.0f + __expf(-x)); }

// ---------------------------------------------------------------------------
// Implicit-GEMM conv: out[b,oc,y,x] = sum_{ic,ky,kx} act[b,ic,y+ky-P,x+kx-P]*W
// act: (64, C_in, 16, 16) fp16 NCHW. wgt: [OC][Kpad] fp16, k = ic*KHW+ky*KW+kx.
// MODE 0: out = f32 NCHW (64, OC, 16,16), bias added.
// MODE 1: ConvTranspose write: row = o*16+dy*4+dx; accumulate into u (64,128,64,64).
// ---------------------------------------------------------------------------
template <int KHW, int KW, int PAD, int MODE>
__launch_bounds__(256)
__global__ void conv_gemm_wmma(const half_t* __restrict__ act,
                               const half_t* __restrict__ wgt,
                               const float*  __restrict__ bias,
                               float* __restrict__ out,
                               int C_in, int OC, int Kpad)
{
    __shared__ half_t As[2][BM * APITCH];     // [buf][oc_local][k]
    __shared__ half_t Bs[2][BN * BPITCH];     // [buf][n_local][k] (transposed)

    const int tid   = threadIdx.x;
    const int wid   = tid >> 5;
    const int lane  = tid & 31;
    const int halfg = lane >> 4;           // K-half group (CDNA5 16-bit layout)
    const int nr    = lane & 15;
    const int n0    = blockIdx.x * BN;     // global column base
    const int oc0   = blockIdx.y * BM;     // global row (output channel) base
    const int m0w   = wid * 16;            // this wave's M sub-tile

    // ---- loop-invariant A-load role: row = tid/2, 16 halfs at col (tid&1)*16
    const int arow   = tid >> 1;
    const int acol   = (tid & 1) * 16;
    const bool avalid = (oc0 + arow) < OC;
    const uint4* asrc = (const uint4*)(wgt + (size_t)(oc0 + arow) * Kpad + acol);

    // ---- loop-invariant B-load role: k = tid&31 fixed; 8 columns, fixed decode
    const int kfix  = tid & 31;
    const int nbase = tid >> 5;
    int bcol_base[8], bcol_py[8], bcol_px[8];
#pragma unroll
    for (int r = 0; r < 8; ++r) {
        int gcol = n0 + nbase + r * 8;
        int bb = gcol >> 8, yx = gcol & 255;
        bcol_base[r] = (bb * C_in) << 8;
        bcol_py[r]   = yx >> 4;
        bcol_px[r]   = yx & 15;
    }

    // ---- register stage ----
    uint4  sa0, sa1;
    half_t sb[8];

    auto load_stage = [&](int kk0) {
        uint4 z = {0u, 0u, 0u, 0u};
        sa0 = z; sa1 = z;
        if (avalid) {
            const uint4* p = asrc + (kk0 >> 3);   // 8 halfs per uint4
            sa0 = p[0];
            sa1 = p[1];
        }
        int gk  = kk0 + kfix;
        int ic  = gk / KHW;                 // compile-time divisor
        int rem = gk - ic * KHW;
        int ky  = rem / KW;
        int kx  = rem - ky * KW;
        bool icv = ic < C_in;
        int icoff = ic << 8;
#pragma unroll
        for (int r = 0; r < 8; ++r) {
            int iy = bcol_py[r] + ky - PAD;
            int ix = bcol_px[r] + kx - PAD;
            half_t val = (half_t)0.0f;
            bool inb = icv;
            if (PAD != 0) inb = inb && (iy >= 0) && (iy < 16) && (ix >= 0) && (ix < 16);
            if (inb)
                val = act[(size_t)(bcol_base[r] + icoff + (iy << 4) + ix)];
            sb[r] = val;
        }
    };
    auto store_stage = [&](int buf) {
        uint4* adst = (uint4*)(&As[buf][arow * APITCH + acol]);
        adst[0] = sa0;
        adst[1] = sa1;
#pragma unroll
        for (int r = 0; r < 8; ++r)
            Bs[buf][(nbase + r * 8) * BPITCH + kfix] = sb[r];
    };

    // ---- accumulators (+bias) ----
    v8f acc[4];
#pragma unroll
    for (int i = 0; i < 4; ++i)
#pragma unroll
        for (int v = 0; v < 8; ++v) acc[i][v] = 0.0f;
    if (MODE == 0 && bias != nullptr) {
#pragma unroll
        for (int v = 0; v < 8; ++v) {
            int oc = oc0 + m0w + v + 8 * halfg;
            float bv = (oc < OC) ? bias[oc] : 0.0f;
#pragma unroll
            for (int i = 0; i < 4; ++i) acc[i][v] = bv;
        }
    }

    // ---- prologue: fill buffer 0 ----
    load_stage(0);
    store_stage(0);
    __syncthreads();

    int cur = 0;
    for (int kk0 = 0; kk0 < Kpad; kk0 += BK) {
        const bool has_next = (kk0 + BK) < Kpad;
        if (has_next) load_stage(kk0 + BK);   // overlaps with WMMAs below

        // ---- load ALL fragments first, then 4 back-to-back WMMAs ----
        union frag_t { uint4 u[2]; v16h v; };
        frag_t afr, bfr[4];
        {
            // A 16x32: lanes 0-15 hold K 0..7 / 16..23; lanes 16-31 K 8..15 / 24..31.
            const uint4* rp = (const uint4*)(&As[cur][(m0w + nr) * APITCH]);
            afr.u[0] = rp[halfg];
            afr.u[1] = rp[2 + halfg];
        }
#pragma unroll
        for (int i = 0; i < 4; ++i) {
            // B 32x16: lanes 0-15 hold K 0..7 / 16..23; lanes 16-31 K 8..15 / 24..31, col=nr.
            const uint4* rp = (const uint4*)(&Bs[cur][(i * 16 + nr) * BPITCH]);
            bfr[i].u[0] = rp[halfg];
            bfr[i].u[1] = rp[2 + halfg];
        }
#pragma unroll
        for (int i = 0; i < 4; ++i)
            acc[i] = __builtin_amdgcn_wmma_f32_16x16x32_f16(
                false, afr.v, false, bfr[i].v, (short)0, acc[i], false, false);

        if (has_next) store_stage(cur ^ 1);   // buffer last read in the previous step
        __syncthreads();
        cur ^= 1;
    }

    // ---- epilogue ----
    if (MODE == 0) {
        const bool full = (oc0 + m0w + 15) < OC;   // wave-uniform fast path
#pragma unroll
        for (int i = 0; i < 4; ++i) {
            int gcol = n0 + i * 16 + nr;
            int bb   = gcol >> 8;
            int yx   = gcol & 255;
#pragma unroll
            for (int v = 0; v < 8; ++v) {
                int row = oc0 + m0w + v + 8 * halfg;
                if (full || row < OC)
                    out[(((size_t)bb * OC + row) << 8) + yx] = acc[i][v];
            }
        }
    } else {
#pragma unroll
        for (int i = 0; i < 4; ++i) {
            int gcol = n0 + i * 16 + nr;
            int bb   = gcol >> 8;
            int yx   = gcol & 255;
            int y = yx >> 4, x = yx & 15;
#pragma unroll
            for (int v = 0; v < 8; ++v) {
                // row = o*16 + dy*4 + dx ; disjoint 4x4 blocks => plain RMW
                int row = oc0 + m0w + v + 8 * halfg;
                int o = row >> 4, dy = (row >> 2) & 3, dx = row & 3;
                size_t ui = (((size_t)bb * 128 + o) * 64 + (y * 4 + dy)) * 64 + (x * 4 + dx);
                out[ui] += acc[i][v];
            }
        }
    }
}

// ---------------------------------------------------------------------------
// Support kernels
// ---------------------------------------------------------------------------
__global__ void pack_weight(const float* __restrict__ w, half_t* __restrict__ out,
                            int OC, int K, int Kpad)
{
    size_t i = (size_t)blockIdx.x * 256 + threadIdx.x;
    size_t total = (size_t)OC * Kpad;
    if (i >= total) return;
    int oc = (int)(i / Kpad);
    int k  = (int)(i - (size_t)oc * Kpad);
    out[i] = (k < K) ? (half_t)w[(size_t)oc * K + k] : (half_t)0.0f;
}

// write_w (in=128, out=128, 4, 4) -> W16[o*16+dy*4+dx][ic]
__global__ void pack_write_w(const float* __restrict__ w, half_t* __restrict__ out)
{
    int i = blockIdx.x * 256 + threadIdx.x;   // 2048*128
    if (i >= 2048 * 128) return;
    int ocp = i >> 7, ic = i & 127;
    int o = ocp >> 4, dy = (ocp >> 2) & 3, dx = ocp & 3;
    out[i] = (half_t)w[(((size_t)ic * 128 + o) * 4 + dy) * 4 + dx];
}

// x (64,3,64,64) conv stride4 4x4 -> fp16 into enc_in channels 128..130
__global__ void read_conv(const float* __restrict__ x, const float* __restrict__ rw,
                          half_t* __restrict__ enc_in)
{
    int i = blockIdx.x * 256 + threadIdx.x;   // 64*3*256
    if (i >= 64 * 3 * 256) return;
    int p = i & 255, t2 = i >> 8;
    int c = t2 % 3, b = t2 / 3;
    int y = p >> 4, xq = p & 15;
    float s = 0.0f;
    for (int ic = 0; ic < 3; ++ic)
        for (int dy = 0; dy < 4; ++dy)
            for (int dx = 0; dx < 4; ++dx)
                s += x[(((size_t)b * 3 + ic) * 64 + (4 * y + dy)) * 64 + (4 * xq + dx)]
                   * rw[((c * 3 + ic) * 4 + dy) * 4 + dx];
    enc_in[((size_t)b * 394 + 128 + c) * 256 + p] = (half_t)s;
}

// broadcast v (64,7) and copy r (64,256,16,16) into both concat buffers (fp16)
__global__ void fill_static(const float* __restrict__ v, const float* __restrict__ r,
                            half_t* __restrict__ enc_in, half_t* __restrict__ dec_in)
{
    size_t i = (size_t)blockIdx.x * 256 + threadIdx.x;  // 64*263*256
    if (i >= (size_t)64 * 263 * 256) return;
    int p  = (int)(i & 255);
    int t2 = (int)(i >> 8);
    int ch = t2 % 263, b = t2 / 263;
    half_t val;
    if (ch < 7) val = (half_t)v[b * 7 + ch];
    else        val = (half_t)r[((size_t)b * 256 + (ch - 7)) * 256 + p];
    enc_in[((size_t)b * 394 + 131 + ch) * 256 + p] = val;
    dec_in[((size_t)b * 327 + 64  + ch) * 256 + p] = val;
}

__global__ void zero_f32(float* p, size_t n)
{ size_t i = (size_t)blockIdx.x * 256 + threadIdx.x; if (i < n) p[i] = 0.0f; }

__global__ void zero_f16(half_t* p, size_t n)
{ size_t i = (size_t)blockIdx.x * 256 + threadIdx.x; if (i < n) p[i] = (half_t)0.0f; }

// tmp16 = base16 + add_f32  (inp = concat + transform(hidden))
__global__ void add_to_h16(const half_t* __restrict__ base, const float* __restrict__ add,
                           half_t* __restrict__ out, size_t n)
{
    size_t i = (size_t)blockIdx.x * 256 + threadIdx.x;
    if (i >= n) return;
    out[i] = (half_t)((float)base[i] + add[i]);
}

// gates (64,512,256) f32 [f,i,o,s] ; c in/out f32 ; h -> fp16 (plus optional
// second fp16 destination with channel offset, used to refresh enc concat h_dec)
__global__ void lstm_update(const float* __restrict__ gates, float* __restrict__ c,
                            half_t* __restrict__ h16, half_t* __restrict__ h16b,
                            int Cb, int ch_off)
{
    int i = blockIdx.x * 256 + threadIdx.x;   // 64*128*256
    if (i >= 64 * 128 * 256) return;
    int p = i & 255, ch = (i >> 8) & 127, b = i >> 15;
    size_t gb = ((size_t)b * 512 + ch) * 256 + p;
    float f  = gates[gb];
    float ig = gates[gb + 128 * 256];
    float o  = gates[gb + 256 * 256];
    float s  = gates[gb + 384 * 256];
    float cv = c[i];
    cv = sigm(f) * cv + sigm(ig) * tanhf(s);
    c[i] = cv;
    float h = sigm(o) * tanhf(cv);
    h16[i] = (half_t)h;
    if (h16b != nullptr)
        h16b[((size_t)b * Cb + ch_off + ch) * 256 + p] = (half_t)h;
}

// z = q_mu + exp(0.5*q_lv)*eps -> dec_in ch 0..63 (fp16); KL accumulation
__global__ void z_kl(const float* __restrict__ prior, const float* __restrict__ post,
                     const float* __restrict__ eps_t, half_t* __restrict__ dec_in,
                     float* __restrict__ kl)
{
    __shared__ float red[256];
    int i = blockIdx.x * 256 + threadIdx.x;   // 64*64*256
    float term = 0.0f;
    if (i < 64 * 64 * 256) {
        int p = i & 255, zc = (i >> 8) & 63, b = i >> 14;
        size_t mu_i = ((size_t)b * 128 + zc) * 256 + p;
        size_t lv_i = ((size_t)b * 128 + 64 + zc) * 256 + p;
        float pmu = prior[mu_i], plv = prior[lv_i];
        float qmu = post[mu_i],  qlv = post[lv_i];
        float z = qmu + __expf(0.5f * qlv) * eps_t[i];
        dec_in[((size_t)b * 327 + zc) * 256 + p] = (half_t)z;
        float d = pmu - qmu;
        term = 0.5f * (__expf(qlv - plv) + d * d * __expf(-plv) - 1.0f + (plv - qlv));
    }
    red[threadIdx.x] = term;
    __syncthreads();
    for (int s = 128; s > 0; s >>= 1) {
        if (threadIdx.x < s) red[threadIdx.x] += red[threadIdx.x + s];
        __syncthreads();
    }
    if (threadIdx.x == 0) atomicAdd(kl, red[0] * (1.0f / 64.0f));   // mean over batch
}

// canvas = sigmoid(1x1 conv(u))  (0.2 GFLOP, VALU is fine)
__global__ void obs_conv(const float* __restrict__ u, const float* __restrict__ ow,
                         const float* __restrict__ ob, float* __restrict__ canvas)
{
    size_t i = (size_t)blockIdx.x * 256 + threadIdx.x;   // 64*3*4096
    if (i >= (size_t)64 * 3 * 4096) return;
    int p  = (int)(i & 4095);
    int t2 = (int)(i >> 12);
    int c = t2 % 3, b = t2 / 3;
    float s = ob[c];
    for (int o = 0; o < 128; ++o)
        s += u[(((size_t)b * 128 + o) << 12) + p] * ow[c * 128 + o];
    canvas[i] = 1.0f / (1.0f + __expf(-s));
}

__global__ void store_kl(const float* __restrict__ kl, float* __restrict__ dst)
{ if (threadIdx.x == 0 && blockIdx.x == 0) *dst = *kl; }

// ---------------------------------------------------------------------------
// Host driver
// ---------------------------------------------------------------------------
static inline size_t align256(size_t x) { return (x + 255) & ~(size_t)255; }

extern "C" void kernel_launch(void* const* d_in, const int* in_sizes, int n_in,
                              void* d_out, int out_size, void* d_ws, size_t ws_size,
                              hipStream_t stream)
{
    (void)in_sizes; (void)n_in; (void)out_size; (void)ws_size;

    const float* x       = (const float*)d_in[0];
    const float* v       = (const float*)d_in[1];
    const float* r       = (const float*)d_in[2];
    const float* eps     = (const float*)d_in[3];
    const float* read_w  = (const float*)d_in[4];
    const float* write_w = (const float*)d_in[5];
    const float* prior_w = (const float*)d_in[6];
    const float* prior_b = (const float*)d_in[7];
    const float* post_w  = (const float*)d_in[8];
    const float* post_b  = (const float*)d_in[9];
    const float* enc_gw  = (const float*)d_in[10];
    const float* enc_gb  = (const float*)d_in[11];
    const float* enc_tw  = (const float*)d_in[12];
    const float* enc_tb  = (const float*)d_in[13];
    const float* dec_gw  = (const float*)d_in[14];
    const float* dec_gb  = (const float*)d_in[15];
    const float* dec_tw  = (const float*)d_in[16];
    const float* dec_tb  = (const float*)d_in[17];
    const float* obs_w   = (const float*)d_in[18];
    const float* obs_b   = (const float*)d_in[19];
    float* out = (float*)d_out;

    // ---- workspace carve-up ----
    char* ws = (char*)d_ws;
    size_t off = 0;
    auto carve = [&](size_t bytes) { char* p = ws + off; off = align256(off + bytes); return p; };

    const int KP_ENC_G = 9856, K_ENC_G = 9850;   // 394*25
    const int KP_DEC_G = 8192, K_DEC_G = 8175;   // 327*25
    const int KP_T     = 3200;                    // 128*25

    half_t* W_enc_g  = (half_t*)carve((size_t)512 * KP_ENC_G * 2);
    half_t* W_enc_t  = (half_t*)carve((size_t)394 * KP_T * 2);
    half_t* W_dec_g  = (half_t*)carve((size_t)512 * KP_DEC_G * 2);
    half_t* W_dec_t  = (half_t*)carve((size_t)327 * KP_T * 2);
    half_t* W_prior  = (half_t*)carve((size_t)128 * KP_T * 2);
    half_t* W_post   = (half_t*)carve((size_t)128 * KP_T * 2);
    half_t* W_write  = (half_t*)carve((size_t)2048 * 128 * 2);
    half_t* enc_in16 = (half_t*)carve((size_t)NB * 394 * SP * 2);
    half_t* dec_in16 = (half_t*)carve((size_t)NB * 327 * SP * 2);
    half_t* tmp_in16 = (half_t*)carve((size_t)NB * 394 * SP * 2);
    half_t* h_enc16  = (half_t*)carve((size_t)NB * 128 * SP * 2);
    half_t* h_dec16  = (half_t*)carve((size_t)NB * 128 * SP * 2);
    float*  c_enc    = (float*)carve((size_t)NB * 128 * SP * 4);
    float*  c_dec    = (float*)carve((size_t)NB * 128 * SP * 4);
    float*  gates    = (float*)carve((size_t)NB * 512 * SP * 4);
    float*  t1f      = (float*)carve((size_t)NB * 394 * SP * 4);
    float*  prior_o  = (float*)carve((size_t)NB * 128 * SP * 4);
    float*  post_o   = (float*)carve((size_t)NB * 128 * SP * 4);
    float*  u        = (float*)carve((size_t)NB * 128 * 64 * 64 * 4);
    float*  klp      = (float*)carve(4);

    auto blocks = [](size_t n) { return (unsigned)((n + 255) / 256); };
    auto grid_for = [](int OC) { return dim3(NTOT / BN, (OC + BM - 1) / BM); };

    // ---- weight packing (every call; deterministic) ----
    pack_weight<<<blocks((size_t)512 * KP_ENC_G), 256, 0, stream>>>(enc_gw, W_enc_g, 512, K_ENC_G, KP_ENC_G);
    pack_weight<<<blocks((size_t)394 * KP_T),     256, 0, stream>>>(enc_tw, W_enc_t, 394, KP_T, KP_T);
    pack_weight<<<blocks((size_t)512 * KP_DEC_G), 256, 0, stream>>>(dec_gw, W_dec_g, 512, K_DEC_G, KP_DEC_G);
    pack_weight<<<blocks((size_t)327 * KP_T),     256, 0, stream>>>(dec_tw, W_dec_t, 327, KP_T, KP_T);
    pack_weight<<<blocks((size_t)128 * KP_T),     256, 0, stream>>>(prior_w, W_prior, 128, KP_T, KP_T);
    pack_weight<<<blocks((size_t)128 * KP_T),     256, 0, stream>>>(post_w,  W_post,  128, KP_T, KP_T);
    pack_write_w<<<blocks((size_t)2048 * 128),    256, 0, stream>>>(write_w, W_write);

    // ---- state init ----
    zero_f16<<<blocks((size_t)NB * 394 * SP), 256, 0, stream>>>(enc_in16, (size_t)NB * 394 * SP);
    zero_f16<<<blocks((size_t)NB * 327 * SP), 256, 0, stream>>>(dec_in16, (size_t)NB * 327 * SP);
    zero_f16<<<blocks((size_t)NB * 128 * SP), 256, 0, stream>>>(h_enc16,  (size_t)NB * 128 * SP);
    zero_f16<<<blocks((size_t)NB * 128 * SP), 256, 0, stream>>>(h_dec16,  (size_t)NB * 128 * SP);
    zero_f32<<<blocks((size_t)NB * 128 * SP), 256, 0, stream>>>(c_enc,    (size_t)NB * 128 * SP);
    zero_f32<<<blocks((size_t)NB * 128 * SP), 256, 0, stream>>>(c_dec,    (size_t)NB * 128 * SP);
    zero_f32<<<blocks((size_t)NB * 128 * 4096), 256, 0, stream>>>(u, (size_t)NB * 128 * 4096);
    zero_f32<<<1, 256, 0, stream>>>(klp, 1);

    // ---- static concat channels ----
    read_conv<<<blocks((size_t)64 * 3 * SP), 256, 0, stream>>>(x, read_w, enc_in16);
    fill_static<<<blocks((size_t)64 * 263 * SP), 256, 0, stream>>>(v, r, enc_in16, dec_in16);

    for (int t = 0; t < 8; ++t) {
        // prior from old h_dec
        conv_gemm_wmma<25,5,2,0><<<grid_for(128), 256, 0, stream>>>(
            h_dec16, W_prior, prior_b, prior_o, 128, 128, KP_T);
        // enc LSTM: inp = concat + transform(h_enc); gates; update
        conv_gemm_wmma<25,5,2,0><<<grid_for(394), 256, 0, stream>>>(
            h_enc16, W_enc_t, enc_tb, t1f, 128, 394, KP_T);
        add_to_h16<<<blocks((size_t)NB * 394 * SP), 256, 0, stream>>>(
            enc_in16, t1f, tmp_in16, (size_t)NB * 394 * SP);
        conv_gemm_wmma<25,5,2,0><<<grid_for(512), 256, 0, stream>>>(
            tmp_in16, W_enc_g, enc_gb, gates, 394, 512, KP_ENC_G);
        lstm_update<<<blocks((size_t)NB * 128 * SP), 256, 0, stream>>>(
            gates, c_enc, h_enc16, (half_t*)nullptr, 0, 0);
        // posterior, z, KL
        conv_gemm_wmma<25,5,2,0><<<grid_for(128), 256, 0, stream>>>(
            h_enc16, W_post, post_b, post_o, 128, 128, KP_T);
        z_kl<<<blocks((size_t)NB * 64 * SP), 256, 0, stream>>>(
            prior_o, post_o, eps + (size_t)t * NB * 64 * SP, dec_in16, klp);
        // dec LSTM (transform uses old h_dec)
        conv_gemm_wmma<25,5,2,0><<<grid_for(327), 256, 0, stream>>>(
            h_dec16, W_dec_t, dec_tb, t1f, 128, 327, KP_T);
        add_to_h16<<<blocks((size_t)NB * 327 * SP), 256, 0, stream>>>(
            dec_in16, t1f, tmp_in16, (size_t)NB * 327 * SP);
        conv_gemm_wmma<25,5,2,0><<<grid_for(512), 256, 0, stream>>>(
            tmp_in16, W_dec_g, dec_gb, gates, 327, 512, KP_DEC_G);
        lstm_update<<<blocks((size_t)NB * 128 * SP), 256, 0, stream>>>(
            gates, c_dec, h_dec16, enc_in16, 394, 0);
        // write: u += ConvTranspose(h_dec) as 1x1 GEMM with scatter epilogue
        conv_gemm_wmma<1,1,0,1><<<grid_for(2048), 256, 0, stream>>>(
            h_dec16, W_write, nullptr, u, 128, 2048, 128);
    }

    obs_conv<<<blocks((size_t)64 * 3 * 4096), 256, 0, stream>>>(u, obs_w, obs_b, out);
    store_kl<<<1, 1, 0, stream>>>(klp, out + (size_t)64 * 3 * 4096);
}
